// AttnTopDown_34626026340889
// MI455X (gfx1250) — matrix operand order repaired and checked by reading, not compiled, not run on hardware
//
#include <hip/hip_runtime.h>
#include <math.h>

// ---------------- types ----------------
typedef __attribute__((ext_vector_type(16))) __bf16 v16bf;
typedef __attribute__((ext_vector_type(8)))  float  v8f;
typedef __attribute__((ext_vector_type(2)))  float  f32x2;
typedef __attribute__((ext_vector_type(2)))  __bf16 bf16x2;
typedef __attribute__((ext_vector_type(4)))  unsigned int u32x4;
typedef __attribute__((ext_vector_type(4)))  int i32x4;
typedef __attribute__((ext_vector_type(8)))  int i32x8;

#define CDIM  256
#define KDIM  256
#define HEADS 8
#define DHEAD 32

#define TM   128
#define TN   128
#define TK   32
#define LDSP 40   // padded LDS stride for B tile (40*2=80 bytes, multiple of 16)

#if __has_builtin(__builtin_amdgcn_tensor_load_to_lds) && __has_builtin(__builtin_amdgcn_s_wait_tensorcnt)
#define USE_TDM 1
#endif

// hardware f32->bf16 (RNE) via convertvector
__device__ __forceinline__ unsigned int cvt2(float lo, float hi) {
  union { bf16x2 h; unsigned int u; } t;
  f32x2 f; f.x = lo; f.y = hi;
  t.h = __builtin_convertvector(f, bf16x2);
  return t.u;
}

__device__ __forceinline__ v16bf load_frag(const unsigned short* p0,
                                           const unsigned short* p1) {
  union { uint4 u[2]; v16bf v; } f;
  f.u[0] = *(const uint4*)p0;
  f.u[1] = *(const uint4*)p1;
  return f.v;
}

// ---------------- weight f32 -> bf16 (per layer, 256x256) ----------------
__global__ __launch_bounds__(256) void cvt_w_kernel(
    const float* __restrict__ w, unsigned short* __restrict__ o)
{
  const int i = (blockIdx.x * 256 + threadIdx.x) * 2;
  *(unsigned int*)&o[i] = cvt2(w[i], w[i + 1]);
}

// ---------------- WMMA GEMM: Y[256 x N] = act( Wbf[256 x 256] * X[256 x N] )
// bf16 weights in global, f32 activations (converted to bf16 en route to LDS),
// f32 accumulate / f32 output. A tile staged by the Tensor Data Mover.
__global__ __launch_bounds__(256) void gemm256_wmma(
    const unsigned short* __restrict__ Wbf, const float* __restrict__ X,
    float* __restrict__ Y, int N, int relu)
{
  __shared__ unsigned short As[TM * TK];    // [m][k] bf16, stride 32 (TDM-landed)
  __shared__ unsigned short Bs[TN * LDSP];  // [n][k] bf16 (transposed, K contiguous)

  const int tid  = threadIdx.x;
  const int lane = tid & 31;
  const int wm   = (tid >> 5) >> 2;   // 0..1 : 64 rows each
  const int wn   = (tid >> 5) & 3;    // 0..3 : 32 cols each
  const int m0   = blockIdx.y * TM;
  const int n0   = blockIdx.x * TN;

  v8f acc[4][2];
#pragma unroll
  for (int i = 0; i < 4; i++)
#pragma unroll
    for (int j = 0; j < 2; j++)
#pragma unroll
      for (int e = 0; e < 8; e++) acc[i][j][e] = 0.0f;

  // B loader: thread covers k-pair (bkrow, bkrow+1) x 8 columns
  const int bkrow = (tid >> 4) * 2;   // 0,2,..,30
  const int bnb   = (tid & 15) * 8;   // 0..120

  const int arow = lane & 15;
  const int ak   = (lane >> 4) * 8;   // A frag: K {ak..ak+7, ak+16..ak+23}
  const int bk   = (lane >> 4) * 8;   // B frag: K {bk..bk+7, bk+16..bk+23}

#ifdef USE_TDM
  // ---- Tensor DMA descriptor (ISA ch.8): 2D tile 32(k) x 128(m) of bf16 ----
  const unsigned long long gbase =
      (unsigned long long)(const void*)(Wbf + (size_t)m0 * KDIM);
  const unsigned int lds_off = (unsigned int)(unsigned long long)(const void*)As;
  u32x4 g0;
  g0.x = 1u;                 // count=1, is_restore=0, gather off
  g0.y = lds_off;            // lds_addr (bytes)
  g0.z = 0u;                 // global_addr lo (per-iteration)
  g0.w = 0u;                 // global_addr hi | type=2 (bits 127:126)
  i32x8 g1;
  g1[0] = 0x00010000;        // workgroup_mask=0, data_size=1 (2 bytes)
  g1[1] = (KDIM << 16);      // tensor_dim0 lo16 in [31:16]
  g1[2] = (256  << 16);      // tensor_dim0 hi16=0, tensor_dim1 lo16=256
  g1[3] = (TK   << 16);      // tensor_dim1 hi16=0, tile_dim0=32
  g1[4] = TM;                // tile_dim1=128, tile_dim2=0
  g1[5] = KDIM;              // tensor_dim0_stride lo32 = 256
  g1[6] = 0;                 // stride hi16, tensor_dim1_stride lo16
  g1[7] = 0;
  i32x4 g2 = {0, 0, 0, 0};
  i32x4 g3 = {0, 0, 0, 0};
  i32x8 g4 = {0, 0, 0, 0, 0, 0, 0, 0};
#else
  const int ar = tid >> 1;            // A row 0..127
  const int ac = (tid & 1) * 16;      // A k-col 0 or 16
#endif

  for (int kk = 0; kk < KDIM; kk += TK) {
    // ---- A tile: rows m0..m0+127, k cols kk..kk+31 (bf16) ----
#ifdef USE_TDM
    {
      unsigned long long ga = gbase + (unsigned long long)(kk * 2);
      g0.z = (unsigned int)ga;
      g0.w = (unsigned int)((ga >> 32) & 0x01ffffffull) | 0x80000000u;
      if (tid < 32)
        __builtin_amdgcn_tensor_load_to_lds(g0, g1, g2, g3, g4, 0);
    }
#else
    {
      const uint4* ap = (const uint4*)(Wbf + (size_t)(m0 + ar) * KDIM + kk + ac);
      uint4 v0 = ap[0], v1 = ap[1];
      uint4* dst = (uint4*)&As[ar * TK + ac];
      dst[0] = v0; dst[1] = v1;
    }
#endif
    // ---- B tile: rows kk+bkrow(+1), cols n0+bnb..+7, f32 -> bf16, transpose
    {
      const float4* r0 = (const float4*)(X + (kk + bkrow) * N + n0 + bnb);
      const float4* r1 = (const float4*)(X + (kk + bkrow + 1) * N + n0 + bnb);
      float4 x0 = r0[0], x1 = r0[1];
      float4 y0 = r1[0], y1 = r1[1];
      float a8[8] = { x0.x, x0.y, x0.z, x0.w, x1.x, x1.y, x1.z, x1.w };
      float b8[8] = { y0.x, y0.y, y0.z, y0.w, y1.x, y1.y, y1.z, y1.w };
#pragma unroll
      for (int i = 0; i < 8; i++)
        *(unsigned int*)&Bs[(bnb + i) * LDSP + bkrow] = cvt2(a8[i], b8[i]);
    }
    if (kk + TK < KDIM)
      __builtin_prefetch(X + (kk + TK + bkrow) * N + n0 + bnb, 0, 0);

#ifdef USE_TDM
    __builtin_amdgcn_s_wait_tensorcnt(0);
#endif
    __syncthreads();

    // ---- fragments (ISA 7.12.2 layouts) + 8 WMMAs ----
    v16bf af[4], bfr[2];
#pragma unroll
    for (int i = 0; i < 4; i++) {
      const int r = wm * 64 + i * 16 + arow;
      af[i] = load_frag(&As[r * TK + ak], &As[r * TK + ak + 16]);
    }
#pragma unroll
    for (int j = 0; j < 2; j++) {
      const int c = wn * 32 + j * 16 + (lane & 15);
      bfr[j] = load_frag(&Bs[c * LDSP + bk], &Bs[c * LDSP + bk + 16]);
    }
#pragma unroll
    for (int i = 0; i < 4; i++)
#pragma unroll
      for (int j = 0; j < 2; j++)
        acc[i][j] = __builtin_amdgcn_wmma_f32_16x16x32_bf16(
            false, af[i], false, bfr[j], (short)0, acc[i][j], false, false);
    __syncthreads();
  }

  // ---- epilogue: D layout = VGPR r: M = base + r + (lane>=16)*8, N = lane&15
#pragma unroll
  for (int i = 0; i < 4; i++) {
    const int row0 = m0 + wm * 64 + i * 16 + (lane >> 4) * 8;
#pragma unroll
    for (int j = 0; j < 2; j++) {
      const int col = n0 + wn * 32 + j * 16 + (lane & 15);
#pragma unroll
      for (int r = 0; r < 8; r++) {
        float vv = acc[i][j][r];
        if (relu) vv = fmaxf(vv, 0.0f);
        Y[(row0 + r) * N + col] = vv;
      }
    }
  }
}

// ---------------- attention pass A: per (h,x,y) softmax weights -----------
__global__ __launch_bounds__(256) void attn_weights_kernel(
    const float* __restrict__ q, const float* __restrict__ k,
    float* __restrict__ w, int Ht, int Wt, int Hb, int Wb)
{
  const int St = Ht * Wt;
  const int Sb = Hb * Wb;
  const int idx = blockIdx.x * blockDim.x + threadIdx.x;
  if (idx >= HEADS * St) return;
  const int xy = idx % St;
  const int h  = idx / St;
  const int x  = xy / Wt;
  const int y  = xy % Wt;

  float kv[DHEAD];
#pragma unroll
  for (int d = 0; d < DHEAD; d++)
    kv[d] = k[(h * DHEAD + d) * St + xy];

  float logits[9];
#pragma unroll
  for (int ki = 0; ki < 3; ki++) {
#pragma unroll
    for (int kj = 0; kj < 3; kj++) {
      const int u = 2 * x + ki - 1;
      const int v = 2 * y + kj - 1;
      float acc = 0.0f;
      if ((unsigned)u < (unsigned)Hb && (unsigned)v < (unsigned)Wb) {
        const float* qp = q + u * Wb + v;
#pragma unroll
        for (int d = 0; d < DHEAD; d++)
          acc = fmaf(qp[(h * DHEAD + d) * Sb], kv[d], acc);
      }
      logits[ki * 3 + kj] = acc * 0.17677669529663687f;  // 1/sqrt(32)
    }
  }
  float m = logits[0];
#pragma unroll
  for (int t = 1; t < 9; t++) m = fmaxf(m, logits[t]);
  float e[9]; float s = 0.0f;
#pragma unroll
  for (int t = 0; t < 9; t++) { e[t] = __expf(logits[t] - m); s += e[t]; }
  const float inv = 1.0f / s;
#pragma unroll
  for (int t = 0; t < 9; t++)
    w[(h * 9 + t) * St + xy] = e[t] * inv;
}

// ---------------- attention pass B: fold collapses to weight-gather -------
__global__ __launch_bounds__(256) void attn_apply_kernel(
    const float* __restrict__ vb, const float* __restrict__ w,
    float* __restrict__ out, int Ht, int Wt, int Hb, int Wb)
{
  const int St = Ht * Wt;
  const int Sb = Hb * Wb;
  const int idx = blockIdx.x * blockDim.x + threadIdx.x;
  if (idx >= CDIM * Sb) return;
  const int sp = idx % Sb;
  const int c  = idx / Sb;
  const int u  = sp / Wb;
  const int vv = sp % Wb;
  const int h  = c / DHEAD;

  float s = 0.0f;
#pragma unroll
  for (int ki = 0; ki < 3; ki++) {
    const int ux = u + 1 - ki;
    if (ux & 1) continue;
    const int x = ux >> 1;
    if ((unsigned)x >= (unsigned)Ht) continue;
#pragma unroll
    for (int kj = 0; kj < 3; kj++) {
      const int vy = vv + 1 - kj;
      if (vy & 1) continue;
      const int y = vy >> 1;
      if ((unsigned)y >= (unsigned)Wt) continue;
      s += w[(h * 9 + ki * 3 + kj) * St + x * Wt + y];
    }
  }
  out[idx] = vb[idx] * s;
}

// ---------------- host orchestration ----------------
extern "C" void kernel_launch(void* const* d_in, const int* in_sizes, int n_in,
                              void* d_out, int out_size, void* d_ws, size_t ws_size,
                              hipStream_t stream)
{
  (void)in_sizes; (void)n_in; (void)out_size; (void)ws_size;
  const float* p3 = (const float*)d_in[0];   // [4,256,128,128]
  const float* p4 = (const float*)d_in[1];   // [4,256, 64, 64]
  const float* p5 = (const float*)d_in[2];   // [4,256, 32, 32]
  const float* Wq[2]  = { (const float*)d_in[3],  (const float*)d_in[8]  };
  const float* Wk[2]  = { (const float*)d_in[4],  (const float*)d_in[9]  };
  const float* Wv[2]  = { (const float*)d_in[5],  (const float*)d_in[10] };
  const float* Ww0[2] = { (const float*)d_in[6],  (const float*)d_in[11] };
  const float* Ww1[2] = { (const float*)d_in[7],  (const float*)d_in[12] };

  float* r3 = (float*)d_out;
  float* r4 = r3 + (size_t)4 * CDIM * 16384;
  float* r5 = r4 + (size_t)4 * CDIM * 4096;

  // per-batch scratch (peak ~56.5 MB); Ww0 output reuses bufQ (q dead by then)
  float* bufQ = (float*)d_ws;              // 256*16384 f32
  float* bufV = bufQ + 4194304;            // 256*16384
  float* bufK = bufV + 4194304;            // 256*4096 (max)
  float* bufO = bufK + 1048576;            // 256*16384
  float* bufW = bufO + 4194304;            // 8*9*4096 (max)
  unsigned short* wbf = (unsigned short*)(bufW + 294912);  // 5 x 65536 bf16

  const float* bots[2] = { p3, p4 };
  const float* tops[2] = { p4, p5 };
  float*       outs[2] = { r3, r4 };
  const int HbA[2] = { 128, 64 };
  const int HtA[2] = { 64, 32 };

  for (int l = 0; l < 2; l++) {
    const int Hb = HbA[l], Wb = Hb, Ht = HtA[l], Wt = Ht;
    const int Sb = Hb * Wb, St = Ht * Wt;
    const dim3 gb(Sb / TN, CDIM / TM);
    const dim3 gt(St / TN, CDIM / TM);

    unsigned short* WqB  = wbf + 0 * 65536;
    unsigned short* WkB  = wbf + 1 * 65536;
    unsigned short* WvB  = wbf + 2 * 65536;
    unsigned short* Ww0B = wbf + 3 * 65536;
    unsigned short* Ww1B = wbf + 4 * 65536;
    cvt_w_kernel<<<128, 256, 0, stream>>>(Wq[l],  WqB);
    cvt_w_kernel<<<128, 256, 0, stream>>>(Wk[l],  WkB);
    cvt_w_kernel<<<128, 256, 0, stream>>>(Wv[l],  WvB);
    cvt_w_kernel<<<128, 256, 0, stream>>>(Ww0[l], Ww0B);
    cvt_w_kernel<<<128, 256, 0, stream>>>(Ww1[l], Ww1B);

    for (int b = 0; b < 4; b++) {
      const float* botb = bots[l] + (size_t)b * CDIM * Sb;
      const float* topb = tops[l] + (size_t)b * CDIM * St;
      float*       outb = outs[l] + (size_t)b * CDIM * Sb;

      gemm256_wmma<<<gb, 256, 0, stream>>>(WqB, botb, bufQ, Sb, 0);
      gemm256_wmma<<<gb, 256, 0, stream>>>(WvB, botb, bufV, Sb, 0);
      gemm256_wmma<<<gt, 256, 0, stream>>>(WkB, topb, bufK, St, 0);

      attn_weights_kernel<<<(HEADS * St + 255) / 256, 256, 0, stream>>>(
          bufQ, bufK, bufW, Ht, Wt, Hb, Wb);
      attn_apply_kernel<<<(CDIM * Sb + 255) / 256, 256, 0, stream>>>(
          bufV, bufW, bufO, Ht, Wt, Hb, Wb);

      gemm256_wmma<<<gb, 256, 0, stream>>>(Ww0B, bufO, bufQ, Sb, 1); // +ReLU
      gemm256_wmma<<<gb, 256, 0, stream>>>(Ww1B, bufQ, outb, Sb, 0);
    }
  }
  // p5 passthrough
  (void)hipMemcpyAsync(r5, p5, (size_t)4 * CDIM * 1024 * sizeof(float),
                       hipMemcpyDeviceToDevice, stream);
}
